// LinearQuantWLog_11201274708565
// MI455X (gfx1250) — matrix operand order repaired and checked by reading, not compiled, not run on hardware
//
#include <hip/hip_runtime.h>

typedef __attribute__((ext_vector_type(16))) _Float16 v16h;
typedef __attribute__((ext_vector_type(8)))  _Float16 v8h;
typedef __attribute__((ext_vector_type(8)))  float    v8f;
typedef __attribute__((ext_vector_type(4)))  float    v4f;
typedef __attribute__((ext_vector_type(4)))  unsigned int v4u;
typedef __attribute__((ext_vector_type(8)))  int      v8i;
typedef __attribute__((ext_vector_type(4)))  int      v4i;

#define IN_F  256
#define OUT_F 256
// LDS row pitch for staged B: 256 halves (512B) + 16B TDM pad = 528B = 264 halves
#define PITCH_H 264
#define SMEM_BYTES (OUT_F * PITCH_H * 2 + 64)   // 135168 + tail pad

// ---------------------------------------------------------------------------
// Kernel 1: quantize fp32 weight [256,256] onto log grid, store f16 to d_ws.
// Grid values: sign(w) * 0.25 * 2^k, k = clip(round(log2(|w|)+2), 0, 4);
// all grid values (+-{0.25,0.5,1,2,4}) are exactly representable in f16.
// ---------------------------------------------------------------------------
__global__ __launch_bounds__(256) void quant_wlog_kernel(
    const float* __restrict__ w, _Float16* __restrict__ wq) {
  int i = blockIdx.x * blockDim.x + threadIdx.x;
  float x = w[i];
  float a = fabsf(x);
  float k = fminf(fmaxf(rintf(log2f(a) + 2.0f), 0.0f), 4.0f);  // RTE = jnp.round
  float q = copysignf(0.25f * exp2f(k), x);
  if (x == 0.0f) q = 0.0f;  // sign(0)==0 in reference
  wq[i] = (_Float16)q;
}

// ---------------------------------------------------------------------------
// Kernel 2: WMMA GEMM, B staged in LDS by the Tensor Data Mover.
// Each wave: 16 token rows x 256 outputs, K=256.  Input split into f16 hi +
// f16 lo (exact two-term split); two v_wmma_f32_16x16x32_f16 per K-chunk.
// ---------------------------------------------------------------------------
__global__ __launch_bounds__(256) void linear_wlog_gemm_kernel(
    const float* __restrict__ x, const _Float16* __restrict__ wq,
    const float* __restrict__ bias, float* __restrict__ out) {
  extern __shared__ _Float16 smem[];   // [256][264] halves, TDM-padded rows

  const int lane  = threadIdx.x & 31;
  const int wave  = threadIdx.x >> 5;
  const int mBase = (blockIdx.x * 8 + wave) * 16;
  const int half  = lane >> 4;   // 0: lanes 0-15, 1: lanes 16-31
  const int l16   = lane & 15;
  const int mRow  = mBase + l16;

  // ---- TDM: DMA the whole quantized weight (256x256 f16) into LDS --------
  // D# group0: count=1 | lds_addr | global_addr | type=2
  // D# group1: data_size=2B, pad_enable, pad_interval=128dw, pad_amount=4dw,
  //            tensor_dim0=256, tensor_dim1=256, tile_dim0=256, tile_dim1=256,
  //            tensor_dim0_stride=256
  if (threadIdx.x == 0) {
    unsigned lds_base = __builtin_amdgcn_groupstaticsize();  // dynamic LDS base
    unsigned long long ga = (unsigned long long)(const void*)wq;
    v4u g0 = {1u,                                     // count=1, gather off
              lds_base,                               // lds_addr [63:32]
              (unsigned)(ga & 0xFFFFFFFFu),           // global_addr lo
              (unsigned)((ga >> 32) & 0x01FFFFFFu) | 0x80000000u};  // hi | type=2
    v8i g1 = {(int)0x07910000,      // data_size=1(2B) | pad_en | intvl=6 | amt=3
              (int)0x01000000,      // tensor_dim0[15:0]=256 <<16 | bar_addr=0
              (int)0x01000000,      // tensor_dim1[15:0]=256 <<16 | tdim0[31:16]
              (int)0x01000000,      // tile_dim0=256 <<16 | tdim1[31:16]
              (int)0x00000100,      // tile_dim2=0 | tile_dim1=256
              (int)0x00000100,      // tensor_dim0_stride=256 (lo 32)
              0,                    // tdim1_stride[15:0] | tdim0_stride[47:32]
              0};                   // tdim1_stride[47:16]
    v4i g2 = {0, 0, 0, 0};          // tensor_dim2.. (unused, 2D tile)
    v4i g3 = {0, 0, 0, 0};          // tensor_dim3_stride.. (unused)
    v8i g4 = {0, 0, 0, 0, 0, 0, 0, 0};  // extension group (unused)
    __builtin_amdgcn_tensor_load_to_lds(g0, g1, g2, g3, g4, 0);
  }

  // ---- Overlap with DMA: preload per-lane bias values --------------------
  float barr[16];
#pragma unroll
  for (int nt = 0; nt < 16; ++nt) barr[nt] = bias[nt * 16 + l16];

  // ---- Overlap with DMA: load A strip (16x256 fp32), split hi/lo f16 -----
  // ISA A-layout (16-bit, 16x32): lane l16 = row M; elements 0..7 hold
  // K = half*8 + 0..7, elements 8..15 hold K = 16 + half*8 + 0..7.
  v16h ahi[8], alo[8];
  const float* xrow = x + (size_t)mRow * IN_F;
#pragma unroll
  for (int kc = 0; kc < 8; ++kc) {
    const int k0 = kc * 32 + half * 8;
    v4f g0 = *(const v4f*)(xrow + k0);
    v4f g1 = *(const v4f*)(xrow + k0 + 4);
    v4f g2 = *(const v4f*)(xrow + k0 + 16);
    v4f g3 = *(const v4f*)(xrow + k0 + 20);
    float g[16] = {g0.x, g0.y, g0.z, g0.w, g1.x, g1.y, g1.z, g1.w,
                   g2.x, g2.y, g2.z, g2.w, g3.x, g3.y, g3.z, g3.w};
    v16h hi, lo;
#pragma unroll
    for (int i = 0; i < 16; ++i) {
      _Float16 h = (_Float16)g[i];          // RTE
      hi[i] = h;
      lo[i] = (_Float16)(g[i] - (float)h);  // exact residual, then rounded
    }
    ahi[kc] = hi;
    alo[kc] = lo;
  }

  // ---- Wait for the tensor DMA, then make LDS visible to all waves -------
  if (threadIdx.x == 0) {
    __builtin_amdgcn_s_wait_tensorcnt(0);
  }
  __syncthreads();

  // ---- 16 output tiles of 16 columns each --------------------------------
#pragma unroll 4
  for (int nt = 0; nt < 16; ++nt) {
    const int col = nt * 16 + l16;
    // B fragment base for this lane: row `col` of staged wq, padded pitch.
    // ISA B-layout (16-bit, 32x16): lane l16 = col N; elements 0..7 hold
    // K = kc*32 + half*8 + 0..7, elements 8..15 hold K = kc*32 + 16 + half*8 + 0..7.
    // 528B pitch => 4-bank shift per row =>
    // the 16 lanes span all 64 LDS banks (conflict-free).
    const _Float16* brow = smem + (size_t)col * PITCH_H + half * 8;
    auto loadB = [&](int kc) -> v16h {
      const _Float16* bp = brow + kc * 32;
      v8h b0 = *(const v8h*)bp;
      v8h b1 = *(const v8h*)(bp + 16);
      v16h b;
#pragma unroll
      for (int i = 0; i < 8; ++i) { b[i] = b0[i]; b[8 + i] = b1[i]; }
      return b;
    };

    // Zero-init both accumulators: first WMMA of each chain takes the
    // inline-constant 0 as C (no register setup, no movrel broadcasts).
    v8f acc_hi = {};
    v8f acc_lo = {};   // two accumulators break the WMMA RAW chain

    // Double-buffered B: issue kc+1 loads before consuming kc, so the
    // s_wait_dscnt overlaps one LDS round-trip with the two WMMAs.
    v16h b = loadB(0);
#pragma unroll
    for (int kc = 0; kc < 8; ++kc) {
      v16h bn = b;
      if (kc < 7) bn = loadB(kc + 1);
      acc_hi = __builtin_amdgcn_wmma_f32_16x16x32_f16(
          false, ahi[kc], false, b, (short)0, acc_hi, false, false);
      acc_lo = __builtin_amdgcn_wmma_f32_16x16x32_f16(
          false, alo[kc], false, b, (short)0, acc_lo, false, false);
      b = bn;
    }

    // Bias added in the tail (scalar broadcast via v_pk_add op_sel).
    const float bv = barr[nt];
    // ISA C/D-layout (32-bit, 16x16): lane l16 = col N; VGPR r = row half*8+r.
    float* orow = out + (size_t)(mBase + half * 8) * OUT_F + col;
#pragma unroll
    for (int r = 0; r < 8; ++r) {
      orow[(size_t)r * OUT_F] = acc_hi[r] + acc_lo[r] + bv;
    }
  }
}

// ---------------------------------------------------------------------------
extern "C" void kernel_launch(void* const* d_in, const int* in_sizes, int n_in,
                              void* d_out, int out_size, void* d_ws, size_t ws_size,
                              hipStream_t stream) {
  const float* input  = (const float*)d_in[0];  // [N, 256] fp32
  const float* weight = (const float*)d_in[1];  // [256, 256] fp32
  const float* bias   = (const float*)d_in[2];  // [256] fp32
  float*       out    = (float*)d_out;          // [N, 256] fp32
  _Float16*    wq     = (_Float16*)d_ws;        // 128 KB quantized weight

  // 1) quantize weight onto log grid (f16-exact values)
  quant_wlog_kernel<<<(OUT_F * IN_F) / 256, 256, 0, stream>>>(weight, wq);

  // 2) WMMA GEMM: 128 token rows per block (8 waves x 16 rows),
  //    B staged in LDS by TDM (132 KB dynamic LDS -> 2 workgroups/WGP)
  const int nTokens = in_sizes[0] / IN_F;       // 262144
  linear_wlog_gemm_kernel<<<nTokens / 128, 256, SMEM_BYTES, stream>>>(
      input, wq, bias, out);
}
